// ExpertChoiceMoE_Fast_52673478918147
// MI455X (gfx1250) — matrix operand order repaired and checked by reading, not compiled, not run on hardware
//
#include <hip/hip_runtime.h>
#include <hip/hip_bf16.h>

// ---------------- problem constants ----------------
#define D_MODEL   1024
#define D_HIDDEN  4096
#define N_EXPERTS 16
#define BT_TOK    32768               // 8 * 4096
#define CAP       2560                // ceil(1.25 * BT / N)
#define TILE_M    16
#define TILES_PER_EXPERT (CAP / TILE_M)   // 160
#define LDA_A     1040                // padded bf16 row stride for x tile
#define LDA_H     4112                // padded bf16 row stride for h tile
// LDS: fp32 staging (64KB) + bf16 A tile + bf16 H tile
#define SMEM_XF32_BYTES (TILE_M * D_MODEL * 4)                     // 65536
#define SMEM_A_BYTES    (TILE_M * LDA_A * 2)                       // 33280
#define SMEM_H_BYTES    (TILE_M * LDA_H * 2)                       // 131584
#define FFN_SMEM_BYTES  (SMEM_XF32_BYTES + SMEM_A_BYTES + SMEM_H_BYTES)

typedef __attribute__((ext_vector_type(16))) __bf16 v16bf;
typedef __attribute__((ext_vector_type(8)))  float  v8f;

#if defined(__gfx1250__) && __has_builtin(__builtin_amdgcn_tensor_load_to_lds)
#define USE_TDM 1
typedef unsigned int tdm_u32x4 __attribute__((ext_vector_type(4)));
typedef int          tdm_i32x8 __attribute__((ext_vector_type(8)));
typedef int          tdm_i32x4 __attribute__((ext_vector_type(4)));
#else
#define USE_TDM 0
#endif

union FragBF {
  v16bf v;
  uint4 q[2];
};

__device__ __forceinline__ int bucket_of(float s) {
  int b = (int)(s * 1024.0f);
  return b > 1023 ? 1023 : (b < 0 ? 0 : b);
}

// branch-free tanh-approx gelu (matches jax.nn.gelu approximate=True form)
__device__ __forceinline__ float gelu_f(float v) {
  float u = 0.7978845608f * (v + 0.044715f * v * v * v);
  float au = __builtin_fabsf(u);
  float ez = __expf(-2.0f * au);
  float th = (1.0f - ez) / (1.0f + ez);
  th = __builtin_copysignf(th, u);
  return 0.5f * v * (1.0f + th);
}

// ---------------- K0: y = x (identity background for scatter) ----------------
__global__ void moe_copy_kernel(const float4* __restrict__ xi,
                                float4* __restrict__ yo, int n) {
  int i = blockIdx.x * blockDim.x + threadIdx.x;
  if (i < n) yo[i] = xi[i];
}

// ---------------- K1: gating: logits -> softmax top1 + histogram -------------
__global__ void moe_gate_kernel(const float* __restrict__ x,
                                const float* __restrict__ gw,
                                float* __restrict__ score,
                                int* __restrict__ eid,
                                int* __restrict__ counts,
                                int* __restrict__ hist) {
  int t = blockIdx.x * blockDim.x + threadIdx.x;
  if (t >= BT_TOK) return;
  const float* xt = x + (size_t)t * D_MODEL;
  float acc[N_EXPERTS];
#pragma unroll
  for (int e = 0; e < N_EXPERTS; ++e) acc[e] = 0.0f;
  for (int d = 0; d < D_MODEL; ++d) {
    float xv = xt[d];
#pragma unroll
    for (int e = 0; e < N_EXPERTS; ++e) acc[e] += xv * gw[e * D_MODEL + d];
  }
  float m = acc[0];
  int best = 0;
#pragma unroll
  for (int e = 1; e < N_EXPERTS; ++e) {
    if (acc[e] > m) { m = acc[e]; best = e; }
  }
  float sum = 0.0f;
#pragma unroll
  for (int e = 0; e < N_EXPERTS; ++e) sum += __expf(acc[e] - m);
  float s = 1.0f / sum;          // softmax max-prob
  score[t] = s;
  eid[t] = best;
  atomicAdd(&counts[best], 1);
  atomicAdd(&hist[best * 1024 + bucket_of(s)], 1);
}

// ---------------- K2: per-expert score threshold + aux losses ----------------
__global__ void moe_select_kernel(const int* __restrict__ counts,
                                  const int* __restrict__ hist,
                                  int* __restrict__ tb,
                                  int* __restrict__ remAllow,
                                  float* __restrict__ out_tail) {
  int tid = threadIdx.x;
  if ((tid & 31) == 0) {
    int e = tid >> 5;
    if (e < N_EXPERTS) {
      int cum = 0, thr = -1, rem = CAP;
      for (int b = 1023; b >= 0; --b) {
        int c = hist[e * 1024 + b];
        if (cum + c >= CAP) { thr = b; rem = CAP - cum; break; }
        cum += c;
      }
      tb[e] = thr;
      remAllow[e] = rem;
    }
  }
  if (tid == 0) {
    const float expected = (float)BT_TOK / (float)N_EXPERTS;  // 2048
    float lb = 0.0f, drop = 0.0f;
    for (int e = 0; e < N_EXPERTS; ++e) {
      float c = (float)counts[e];
      float d = c - expected;
      lb += d * d;
      float ov = c - (float)CAP;
      if (ov > 0.0f) drop += ov;
    }
    out_tail[0] = (lb / (float)N_EXPERTS) / (expected * expected);
    out_tail[1] = drop / (float)BT_TOK;
  }
}

// ---------------- K3: slot assignment per token ----------------
__global__ void moe_assign_kernel(const float* __restrict__ score,
                                  const int* __restrict__ eid,
                                  const int* __restrict__ tb,
                                  const int* __restrict__ remAllow,
                                  int* __restrict__ slotCnt,
                                  int* __restrict__ tieCnt,
                                  int* __restrict__ rowTok,
                                  int* __restrict__ rowValid) {
  int t = blockIdx.x * blockDim.x + threadIdx.x;
  if (t >= BT_TOK) return;
  int e = eid[t];
  int b = bucket_of(score[t]);
  int thr = tb[e];
  bool take = false;
  if (thr < 0 || b > thr) {
    take = true;
  } else if (b == thr) {
    int r = atomicAdd(&tieCnt[e], 1);
    if (r < remAllow[e]) take = true;
  }
  if (take) {
    int slot = atomicAdd(&slotCnt[e], 1);
    if (slot < CAP) {
      rowTok[e * CAP + slot] = t;
      rowValid[e * CAP + slot] = 1;
    }
  }
}

// ---------------- K4: fused expert FFN (WMMA bf16, h tile in LDS) ------------
__global__ __launch_bounds__(256) void moe_ffn_kernel(
    const float* __restrict__ x,
    const float* __restrict__ w1, const float* __restrict__ b1,
    const float* __restrict__ w2, const float* __restrict__ b2,
    const int* __restrict__ rowTok, const int* __restrict__ rowValid,
    float* __restrict__ y) {
  extern __shared__ __align__(16) char smem[];
  float*  sXf32 = (float*)smem;                                   // [16][1024] fp32
  __bf16* sA = (__bf16*)(smem + SMEM_XF32_BYTES);                 // [16][LDA_A]
  __bf16* sH = (__bf16*)(smem + SMEM_XF32_BYTES + SMEM_A_BYTES);  // [16][LDA_H]
  __shared__ int sTok[TILE_M];
  __shared__ int sVal[TILE_M];

  const int e    = blockIdx.y;
  const int tile = blockIdx.x;
  const int tid  = threadIdx.x;
  const int lane = tid & 31;
  const int wave = tid >> 5;

  if (tid < TILE_M) {
    int r = e * CAP + tile * TILE_M + tid;
    int tok = rowTok[r];
    sTok[tid] = (tok < 0 || tok >= BT_TOK) ? 0 : tok;
    sVal[tid] = rowValid[r];
  }
  __syncthreads();

#if USE_TDM
  // TDM gather: one DMA pulls the 16 scattered fp32 token rows into LDS.
  if (wave == 0) {
    unsigned long long gaddr = (unsigned long long)(const void*)x;
    unsigned lds_off = (unsigned)(unsigned long long)(void*)sXf32;
    tdm_u32x4 g0;
    g0.x = 0x80000001u;  // gather_mode=1, 16-bit indices, count=1
    g0.y = lds_off;
    g0.z = (unsigned)(gaddr & 0xFFFFFFFFull);
    g0.w = (unsigned)((gaddr >> 32) & 0x01FFFFFFull) | (2u << 30);  // type=2
    tdm_i32x8 g1;
    g1[0] = 0x00020000;             // data_size=2 (4-byte elements)
    g1[1] = (int)(1024u << 16);     // tensor_dim0 = 1024 (low 16 bits)
    g1[2] = (int)0x80000000u;       // tensor_dim1 = 32768 rows (low 16 bits)
    g1[3] = (int)(1024u << 16);     // tile_dim0 = 1024
    g1[4] = 16;                     // tile_dim1 = number of gather indices
    g1[5] = 1024;                   // tensor_dim0_stride = 1024 elements
    g1[6] = 0;
    g1[7] = 0;
    tdm_i32x4 g2, g3;
#pragma unroll
    for (int k = 0; k < 4; ++k) {
      g2[k] = (sTok[2 * k] & 0xFFFF) | (sTok[2 * k + 1] << 16);
      g3[k] = (sTok[8 + 2 * k] & 0xFFFF) | (sTok[8 + 2 * k + 1] << 16);
    }
#if __clang_major__ >= 23
    tdm_i32x8 gz = {};
    __builtin_amdgcn_tensor_load_to_lds(g0, g1, g2, g3, gz, 0);
#else
    __builtin_amdgcn_tensor_load_to_lds(g0, g1, g2, g3, 0);
#endif
#if __has_builtin(__builtin_amdgcn_s_wait_tensorcnt)
    __builtin_amdgcn_s_wait_tensorcnt(0);
#else
    asm volatile("s_wait_tensorcnt 0x0" ::: "memory");
#endif
  }
  __syncthreads();
  for (int i = tid; i < TILE_M * D_MODEL; i += 256) {
    int r = i >> 10;
    int d = i & (D_MODEL - 1);
    sA[r * LDA_A + d] = (__bf16)sXf32[(i)];
    (void)d;
  }
#else
  // Fallback: direct per-thread gather + convert
  for (int i = tid; i < TILE_M * D_MODEL; i += 256) {
    int r = i >> 10;
    int d = i & (D_MODEL - 1);
    sA[r * LDA_A + d] = (__bf16)x[(size_t)sTok[r] * D_MODEL + d];
  }
#endif
  __syncthreads();

  // Hoist per-lane routing state out of the hot loops.
  int vmask = 0;
#pragma unroll
  for (int m = 0; m < TILE_M; ++m) vmask |= (sVal[m] ? (1 << m) : 0);
  const int row   = lane & 15;   // M for A/C, N-within-tile for B/C
  const int half  = lane >> 4;   // which K half this lane holds
  const int koff  = half * 16;   // A fragment K sub-offset (contiguous K, matches B)
  const int kbrow = half * 16;   // B fragment K sub-offset (contiguous K, matches A)
  int myTok[8];
#pragma unroll
  for (int v = 0; v < 8; ++v) myTok[v] = sTok[half * 8 + v];

  const float* w1e = w1 + (size_t)e * D_MODEL * D_HIDDEN;
  const float* b1e = b1 + (size_t)e * D_HIDDEN;

  // ---- GEMM1: h[16,4096] = gelu(A[16,1024] * W1 + b1); wave owns 512 cols ---
  for (int nt = 0; nt < 32; ++nt) {
    const int ncol = wave * 512 + nt * 16;
    const int nc = ncol + row;
    v8f c = {};
#pragma unroll 2
    for (int kb = 0; kb < D_MODEL; kb += 32) {
      FragBF a, b;
      a.q[0] = *(const uint4*)(sA + row * LDA_A + kb + koff);
      a.q[1] = *(const uint4*)(sA + row * LDA_A + kb + 8 + koff);
      const float* bp = w1e + (size_t)(kb + kbrow) * D_HIDDEN + nc;
      __builtin_prefetch(bp + (size_t)32 * D_HIDDEN, 0, 1);
#pragma unroll
      for (int j = 0; j < 16; ++j) b.v[j] = (__bf16)bp[(size_t)j * D_HIDDEN];
      c = __builtin_amdgcn_wmma_f32_16x16x32_bf16(false, a.v, false, b.v,
                                                  (short)0, c, false, false);
    }
    const float bias = b1e[nc];
#pragma unroll
    for (int v = 0; v < 8; ++v) {
      int m = half * 8 + v;   // C-layout: lanes16-31 hold M=8..15
      sH[m * LDA_H + nc] = (__bf16)gelu_f(c[v] + bias);
    }
  }
  __syncthreads();

  // ---- GEMM2: ye[16,1024] = h[16,4096] * W2 + b2; wave owns 128 cols --------
  const float* w2e = w2 + (size_t)e * D_HIDDEN * D_MODEL;
  const float* b2e = b2 + (size_t)e * D_MODEL;
  for (int nt = 0; nt < 8; ++nt) {
    const int ncol = wave * 128 + nt * 16;
    const int nc = ncol + row;
    v8f c = {};
#pragma unroll 2
    for (int kb = 0; kb < D_HIDDEN; kb += 32) {
      FragBF a, b;
      a.q[0] = *(const uint4*)(sH + row * LDA_H + kb + koff);
      a.q[1] = *(const uint4*)(sH + row * LDA_H + kb + 8 + koff);
      const float* bp = w2e + (size_t)(kb + kbrow) * D_MODEL + nc;
      __builtin_prefetch(bp + (size_t)32 * D_MODEL, 0, 1);
#pragma unroll
      for (int j = 0; j < 16; ++j) b.v[j] = (__bf16)bp[(size_t)j * D_MODEL];
      c = __builtin_amdgcn_wmma_f32_16x16x32_bf16(false, a.v, false, b.v,
                                                  (short)0, c, false, false);
    }
    const float bias = b2e[nc];
#pragma unroll
    for (int v = 0; v < 8; ++v) {
      int m = half * 8 + v;
      if (vmask & (1 << m)) {
        y[(size_t)myTok[v] * D_MODEL + nc] = c[v] + bias;  // unique per token
      }
    }
  }
}

// ---------------- host launcher ----------------
extern "C" void kernel_launch(void* const* d_in, const int* in_sizes, int n_in,
                              void* d_out, int out_size, void* d_ws, size_t ws_size,
                              hipStream_t stream) {
  const float* x      = (const float*)d_in[0];
  const float* gate_w = (const float*)d_in[1];
  const float* w1     = (const float*)d_in[2];
  const float* b1     = (const float*)d_in[3];
  const float* w2     = (const float*)d_in[4];
  const float* b2     = (const float*)d_in[5];
  float* out = (float*)d_out;

  // workspace partition
  char* p = (char*)d_ws;
  float* score = (float*)p;                 p += (size_t)BT_TOK * sizeof(float);
  int*   eid   = (int*)p;                   p += (size_t)BT_TOK * sizeof(int);
  int*   ctrl  = (int*)p;
  int* counts  = ctrl;                        // N
  int* hist    = counts + N_EXPERTS;          // N * 1024
  int* tb      = hist + N_EXPERTS * 1024;     // N
  int* remA    = tb + N_EXPERTS;              // N
  int* slotC   = remA + N_EXPERTS;            // N
  int* tieC    = slotC + N_EXPERTS;           // N
  int* rowTok  = tieC + N_EXPERTS;            // N * CAP
  int* rowVal  = rowTok + N_EXPERTS * CAP;    // N * CAP
  size_t ctrlInts = (size_t)N_EXPERTS * (1024 + 5) + 2ull * N_EXPERTS * CAP;
  hipMemsetAsync(ctrl, 0, ctrlInts * sizeof(int), stream);

  // y = x background
  int n4 = BT_TOK * D_MODEL / 4;
  moe_copy_kernel<<<n4 / 256, 256, 0, stream>>>((const float4*)x, (float4*)out, n4);

  // gating / routing
  moe_gate_kernel<<<BT_TOK / 256, 256, 0, stream>>>(x, gate_w, score, eid, counts, hist);
  moe_select_kernel<<<1, 512, 0, stream>>>(counts, hist, tb, remA,
                                           out + (size_t)BT_TOK * D_MODEL);
  moe_assign_kernel<<<BT_TOK / 256, 256, 0, stream>>>(score, eid, tb, remA,
                                                      slotC, tieC, rowTok, rowVal);

  // fused expert FFN
  dim3 grid(TILES_PER_EXPERT, N_EXPERTS);
  moe_ffn_kernel<<<grid, 256, FFN_SMEM_BYTES, stream>>>(x, w1, b1, w2, b2,
                                                        rowTok, rowVal, out);
}
